// ScaledDotProductAttention_7748121002012
// MI455X (gfx1250) — matrix operand order repaired and checked by reading, not compiled, not run on hardware
//
#include <hip/hip_runtime.h>

typedef __attribute__((ext_vector_type(16))) _Float16 v16h;
typedef __attribute__((ext_vector_type(8)))  _Float16 v8h;
typedef __attribute__((ext_vector_type(8)))  float    v8f;
typedef __attribute__((ext_vector_type(4)))  float    v4f;

#define B_ 2
#define H_ 16
#define S_ 2048
#define D_ 64
#define SCALE_ 0.125f        /* 1/sqrt(64), folded into Q fragments */
#define NEG_BIG (-1e9f)

// --- cross-lane row reductions within a 16-lane half (one C-fragment row) ---
static __device__ __forceinline__ float hmax16(float v) {
#pragma unroll
    for (int off = 1; off < 16; off <<= 1)
        v = fmaxf(v, __shfl_xor(v, off, 16));
    return v;
}
static __device__ __forceinline__ float hsum16(float v) {
#pragma unroll
    for (int off = 1; off < 16; off <<= 1)
        v += __shfl_xor(v, off, 16);
    return v;
}

// two 8-fp32 runs p[0..8) and p[16..24) -> 16 f16 fragment elements (4x b128 loads + pk cvt)
static __device__ __forceinline__ v16h cvt16(const float* __restrict__ p) {
    const v4f* q = (const v4f*)p;
    v16h h;
#pragma unroll
    for (int i = 0; i < 4; ++i) {
        v4f f = q[(i & 1) + (i >> 1) * 4];
#pragma unroll
        for (int j = 0; j < 4; ++j) h[4 * i + j] = (_Float16)f[j];
    }
    return h;
}

// two contiguous 8-fp32 runs -> A-fragment, scaled by SCALE_
static __device__ __forceinline__ v16h cvt_q(const float* __restrict__ p0,
                                             const float* __restrict__ p1) {
    const v4f* a = (const v4f*)p0;
    const v4f* b = (const v4f*)p1;
    v16h h;
#pragma unroll
    for (int i = 0; i < 2; ++i) {
        v4f f0 = a[i], f1 = b[i];
#pragma unroll
        for (int j = 0; j < 4; ++j) {
            h[4 * i + j]     = (_Float16)(f0[j] * SCALE_);
            h[8 + 4 * i + j] = (_Float16)(f1[j] * SCALE_);
        }
    }
    return h;
}

// S_tile(16q x 16k) = Qs(16x64) * K^T(64x16), two v_wmma_f32_16x16x32_f16.
// B-layout (32x16 f16): lane n holds col n; its 16 elements are the feature
// dims [32c+8*hi, +8) and [32c+16+8*hi, +8) of K-row kk0+n.
static __device__ __forceinline__ v8f score_tile(const v16h qa[2],
                                                 const float* __restrict__ Kb,
                                                 int kk0, int n, int hi) {
    v8f acc = {0.f, 0.f, 0.f, 0.f, 0.f, 0.f, 0.f, 0.f};
#pragma unroll
    for (int c = 0; c < 2; ++c) {
        v16h kb = cvt16(Kb + (size_t)(kk0 + n) * D_ + c * 32 + 8 * hi);
        acc = __builtin_amdgcn_wmma_f32_16x16x32_f16(false, qa[c], false, kb,
                                                     (short)0, acc, false, false);
    }
    return acc;
}

__global__ __launch_bounds__(256) void attn_fwd_wmma(
    const float* __restrict__ Q, const float* __restrict__ K,
    const float* __restrict__ V, float* __restrict__ ctx,
    float* __restrict__ prob) {
    // per-wave staging tile: P C-fragment -> A-fragment relayout (1KB/wave)
    __shared__ __align__(16) _Float16 pstage[8][16 * 32];

    const int lane = threadIdx.x & 31;
    const int wave = threadIdx.x >> 5;
    const int n    = lane & 15;             // C/B fragment column
    const int m    = lane & 15;             // A fragment row
    const int hi   = lane >> 4;             // half-wave select
    const int qt   = blockIdx.x * 8 + wave; // 16-row query tile id (0..127)
    const int bh   = blockIdx.y;            // fused batch*head (0..31)
    const int q0   = qt * 16;
    const int qmax = q0 + 15;

    const float* Qb = Q + (size_t)bh * S_ * D_;
    const float* Kb = K + (size_t)bh * S_ * D_;
    const float* Vb = V + (size_t)bh * S_ * D_;
    float* Cb = ctx  + (size_t)bh * S_ * D_;
    float* Pb = prob + (size_t)bh * S_ * S_;

    // ---- Q tile once, pre-scaled, as two 16x32 f16 A-fragments ----
    v16h qa[2];
#pragma unroll
    for (int c = 0; c < 2; ++c) {
        const float* base = Qb + (size_t)(q0 + m) * D_ + c * 32 + 8 * hi;
        qa[c] = cvt_q(base, base + 16);
    }

    // ---- Phase 1: online (rowmax, rowsum) over causal 32-key blocks ----
    float mrow[8], srow[8];
#pragma unroll
    for (int r = 0; r < 8; ++r) { mrow[r] = -__builtin_inff(); srow[r] = 0.f; }

    const int nb = ((16 * qt + 15) >> 5) + 1;  // # causal 32-key blocks
    for (int b = 0; b < nb; ++b) {
        const int kk0 = b << 5;
        v8f sc0 = score_tile(qa, Kb, kk0, n, hi);
        v8f sc1 = score_tile(qa, Kb, kk0 + 16, n, hi);
        if (b + 1 < nb)
            __builtin_prefetch(Kb + (size_t)(kk0 + 32 + n) * D_, 0, 1);
        const bool needMask = (kk0 + 31) > q0;  // block touches the diagonal
#pragma unroll
        for (int r = 0; r < 8; ++r) {
            const int row = q0 + r + 8 * hi;
            float sv0 = sc0[r];
            float sv1 = sc1[r];
            if (needMask) {
                if (kk0 + n > row)      sv0 = NEG_BIG;
                if (kk0 + 16 + n > row) sv1 = NEG_BIG;
            }
            const float nm = fmaxf(mrow[r], hmax16(fmaxf(sv0, sv1)));
            const float ts = hsum16(__expf(sv0 - nm) + __expf(sv1 - nm));
            srow[r] = srow[r] * __expf(mrow[r] - nm) + ts;
            mrow[r] = nm;
        }
    }
    float sinv[8];
#pragma unroll
    for (int r = 0; r < 8; ++r) sinv[r] = 1.0f / srow[r];

    // ---- Phase 2: normalized probs (streaming NT stores) + P*V ----
    v8f acc[4];
#pragma unroll
    for (int dn = 0; dn < 4; ++dn) {
        v8f z = {0.f, 0.f, 0.f, 0.f, 0.f, 0.f, 0.f, 0.f};
        acc[dn] = z;
    }
    _Float16* ps = pstage[wave];

    for (int kb32 = 0; kb32 < S_ / 32; ++kb32) {
        const int kk0 = kb32 << 5;
        if (kk0 <= qmax) {
            const bool needMask = (kk0 + 31) > q0;
            if (kk0 + 32 <= qmax)
                __builtin_prefetch(Vb + (size_t)(kk0 + 32 + n) * D_, 0, 1);
#pragma unroll
            for (int t = 0; t < 2; ++t) {
                const int k0t = kk0 + t * 16;
                v8f sc = score_tile(qa, Kb, k0t, n, hi);
#pragma unroll
                for (int r = 0; r < 8; ++r) {
                    const int lr  = r + 8 * hi;
                    const int row = q0 + lr;
                    float sv = sc[r];
                    if (needMask && (k0t + n > row)) sv = NEG_BIG;
                    const float p = __expf(sv - mrow[r]) * sinv[r];
                    __builtin_nontemporal_store(p,
                        Pb + (size_t)row * S_ + k0t + n);
                    ps[lr * 32 + t * 16 + n] = (_Float16)p;
                }
            }
            // P C-fragment -> A-fragment via LDS (2x ds_load_b128)
            v16h pa;
            {
                v8h lo = *(const v8h*)(ps + m * 32 + 16 * hi);
                v8h hh = *(const v8h*)(ps + m * 32 + 8 + 16 * hi);
#pragma unroll
                for (int i = 0; i < 8; ++i) { pa[i] = lo[i]; pa[8 + i] = hh[i]; }
            }
            // context(16x64) += P(16x32) * V(32x64), 4 N-chunks of 16
#pragma unroll
            for (int dn = 0; dn < 4; ++dn) {
                v16h vb;
#pragma unroll
                for (int v = 0; v < 8; ++v) {
                    const int kd = 2 * v + 16 * hi;
                    const float* p =
                        Vb + (size_t)(kk0 + kd) * D_ + dn * 16 + n;
                    vb[2 * v]     = (_Float16)p[0];
                    vb[2 * v + 1] = (_Float16)p[D_];
                }
                acc[dn] = __builtin_amdgcn_wmma_f32_16x16x32_f16(
                    false, pa, false, vb, (short)0, acc[dn], false, false);
            }
        } else {
            // fully masked block: exact zeros, b128 NT streaming stores
#pragma unroll
            for (int it = 0; it < 4; ++it) {
                const int idx = it * 32 + lane;        // 128 v4f chunks / 4 iters
                const int row = q0 + (idx >> 3);
                const int c4  = (idx & 7) * 4;
                v4f z = {0.f, 0.f, 0.f, 0.f};
                __builtin_nontemporal_store(
                    z, (v4f*)(Pb + (size_t)row * S_ + kk0 + c4));
            }
        }
    }

    // ---- context tile (P was normalized before the GEMM) ----
#pragma unroll
    for (int dn = 0; dn < 4; ++dn)
#pragma unroll
        for (int r = 0; r < 8; ++r)
            Cb[(size_t)(q0 + r + 8 * hi) * D_ + dn * 16 + n] = acc[dn][r];
}

extern "C" void kernel_launch(void* const* d_in, const int* in_sizes, int n_in,
                              void* d_out, int out_size, void* d_ws, size_t ws_size,
                              hipStream_t stream) {
    const float* Q = (const float*)d_in[0];
    const float* K = (const float*)d_in[1];
    const float* V = (const float*)d_in[2];
    // d_in[3] = attn_mask (bool causal) — causality applied analytically.
    float* ctx  = (float*)d_out;
    float* prob = ctx + (size_t)B_ * H_ * S_ * D_;  // tuple order: (context, attn_prob)

    dim3 grid(S_ / (16 * 8), B_ * H_);  // (16, 32): 8 q-tile waves per block
    dim3 block(256);
    hipLaunchKernelGGL(attn_fwd_wmma, grid, block, 0, stream, Q, K, V, ctx, prob);
}
